// Dconv_drop_21827023798972
// MI455X (gfx1250) — matrix operand order repaired and checked by reading, not compiled, not run on hardware
//
#include <hip/hip_runtime.h>

typedef __attribute__((ext_vector_type(16))) __bf16 v16bf;
typedef __attribute__((ext_vector_type(8)))  float  v8f;

namespace {
constexpr int H_    = 56;
constexpr int W_    = 56;
constexpr int HW_   = H_ * W_;        // 3136
constexpr int CIN_  = 64;
constexpr int COUT_ = 64;
constexpr int KK_   = 9;              // 3x3 taps
constexpr int KTOT_ = CIN_ * KK_;     // 576 = GEMM K
constexpr int B_    = 32;
constexpr int NT_   = 64;             // output pixels per workgroup
constexpr int NCH_  = KTOT_ / 32;     // 18 K-chunks of 32
// workspace layout: [0, WPK) weight B-fragments ; [WPK, WPK+XT) transposed bf16 x
constexpr size_t WPK_BYTES = (size_t)NCH_ * 4 * 32 * 32;        // 73728
constexpr size_t XT_BYTES  = (size_t)B_ * HW_ * CIN_ * 2;       // 12845056
}

// ---------- one-shot: transpose x (b,c,p) f32 -> xt (b,p,c) bf16 ----------
__global__ __launch_bounds__(256)
void transpose_x_bf16(const float* __restrict__ x, unsigned short* __restrict__ xt)
{
  __shared__ unsigned short tile[64][66];     // [c][p], padded
  const int t  = threadIdx.x;
  const int b  = blockIdx.y;
  const int p0 = blockIdx.x * 64;
  const float* xb = x + (size_t)b * CIN_ * HW_ + p0;

#pragma unroll
  for (int it = 0; it < 16; ++it) {           // 64c x 64p, reads coalesced along p
    const int idx = it * 256 + t;
    const int c = idx >> 6, p = idx & 63;
    union { __bf16 h; unsigned short u; } cv;
    cv.h = (__bf16)xb[(size_t)c * HW_ + p];
    tile[c][p] = cv.u;
  }
  __syncthreads();

  const int p  = t >> 2;                      // writes coalesced along c
  const int cg = (t & 3) * 16;
  union { unsigned short u[16]; uint4 q[2]; } R;
#pragma unroll
  for (int i = 0; i < 16; ++i) R.u[i] = tile[cg + i][p];
  uint4* dst = (uint4*)(xt + ((size_t)(b * HW_ + p0 + p) * CIN_ + cg));
  dst[0] = R.q[0];
  dst[1] = R.q[1];
}

// ---------- one-shot: pack weights as WMMA B-fragments, K' = kk*64 + c ----------
// chunk q: kk = q/2, channel half cb = (q%2)*32.  Fragment (q, n, lane):
//   lane <-> N (o = n*16+lane%16), elements e <-> K (c = cb + table order of (lane/16, e))
__global__ __launch_bounds__(256)
void prepack_w_bfrag(const float* __restrict__ wgt, uint4* __restrict__ wpk)
{
  const int gid = blockIdx.x * 256 + threadIdx.x;
  if (gid >= NCH_ * 4 * 32) return;
  const int l  = gid & 31;
  const int n  = (gid >> 5) & 3;
  const int q  = gid >> 7;
  const int kk = q >> 1;
  const int cb = (q & 1) * 32;
  union { __bf16 h[16]; uint4 v[2]; } U;
#pragma unroll
  for (int e = 0; e < 16; ++e)
    U.h[e] = (__bf16)wgt[(size_t)(n * 16 + (l & 15)) * KTOT_ + (cb + (e < 8 ? 8 * (l >> 4) + e : 8 + 8 * (l >> 4) + e)) * KK_ + kk];
  wpk[(size_t)gid * 2]     = U.v[0];
  wpk[(size_t)gid * 2 + 1] = U.v[1];
}

// ---------- main kernel: no LDS, no barriers, software-pipelined operand loads ----------
// M = pixels (4 x 16 per WG), N = output channels (4 x 16), 4 waves, 4 acc/wave.
__global__ __launch_bounds__(128)
void dconv_drop_wmma_t(const unsigned short* __restrict__ xt,  // (b, p, c) bf16
                       const uint4* __restrict__ wpk,
                       const int*   __restrict__ sidx,
                       float* __restrict__ out)
{
  const int t    = threadIdx.x;
  const int lane = t & 31;
  const int m    = t >> 5;                       // wave = pixel sub-tile
  const int b    = blockIdx.y;
  const int pix0 = blockIdx.x * NT_;

  const int prow = pix0 + m * 16 + (lane & 15);  // this lane's A-matrix row (pixel)
  const int offb = (lane < 16) ? 0 : 16;         // lane-half channel offset (8ch * 2B)

  // 9 tap row offsets into xt (bytes), lane-half folded in -> zero addr VALU in loop
  const unsigned xtb = (unsigned)b * (unsigned)(HW_ * CIN_ * 2);
  unsigned ro[KK_];
#pragma unroll
  for (int kk = 0; kk < KK_; ++kk)
    ro[kk] = xtb + ((unsigned)sidx[prow * KK_ + kk] << 7) + offb;

  const char* xtc = (const char*)xt;
  const char* wpc = (const char*)wpk + (unsigned)lane * 32;

  // double-buffered operand registers (compile-time indexed after full unroll)
  uint4 Abuf[2][4];      // [ping][offsets 0,32,64,96] for one kk (both channel halves)
  uint4 Bbuf[2][4][2];   // [ping][n][2 x 16B]

  auto loadA = [&](int kk, int pp) {
    const char* p = xtc + ro[kk];
    Abuf[pp][0] = *(const uint4*)(p);
    Abuf[pp][1] = *(const uint4*)(p + 32);
    Abuf[pp][2] = *(const uint4*)(p + 64);
    Abuf[pp][3] = *(const uint4*)(p + 96);
  };
  auto loadB = [&](int q, int pp) {
    const char* p = wpc + q * 4096;
#pragma unroll
    for (int n = 0; n < 4; ++n) {
      Bbuf[pp][n][0] = *(const uint4*)(p + n * 1024);
      Bbuf[pp][n][1] = *(const uint4*)(p + n * 1024 + 16);
    }
  };

  v8f acc[4] = {};
  loadA(0, 0);
  loadB(0, 0);

#pragma unroll
  for (int q = 0; q < NCH_; ++q) {
    const int pb = q & 1;                        // B ping-pong
    const int pa = (q >> 1) & 1;                 // A ping-pong (per kk)

    if (q + 1 < NCH_) loadB(q + 1, pb ^ 1);      // next chunk's weights in flight
    if (pb == 0 && q + 2 < NCH_) loadA((q >> 1) + 1, pa ^ 1);  // next tap's pixels

    union { v16bf v; uint4 u[2]; } A;
    A.u[0] = Abuf[pa][pb * 2 + 0];               // c = cb+off .. +7
    A.u[1] = Abuf[pa][pb * 2 + 1];               // c = cb+16+off .. +23
#pragma unroll
    for (int n = 0; n < 4; ++n) {
      union { v16bf v; uint4 u[2]; } Bf;
      Bf.u[0] = Bbuf[pb][n][0];
      Bf.u[1] = Bbuf[pb][n][1];
      acc[n] = __builtin_amdgcn_wmma_f32_16x16x32_bf16(false, A.v, false, Bf.v,
                                                       (short)0, acc[n], false, false);
    }
  }

  // C tile: lane = N (output channel), vgpr j = M (pixel, +8 for high lanes)
  const int ocol = lane & 15;
  const int moff = (lane < 16) ? 0 : 8;
  float* ob = out + (size_t)b * COUT_ * HW_ + pix0 + m * 16 + moff;
#pragma unroll
  for (int n = 0; n < 4; ++n) {
    float* op = ob + (size_t)(n * 16 + ocol) * HW_;
    float4 lo = { acc[n][0], acc[n][1], acc[n][2], acc[n][3] };
    float4 hi = { acc[n][4], acc[n][5], acc[n][6], acc[n][7] };
    *reinterpret_cast<float4*>(op)     = lo;     // 8 consecutive pixels per lane
    *reinterpret_cast<float4*>(op + 4) = hi;
  }
}

// ---------- self-contained fallback (tiny ws): LDS-staged path ----------
__global__ __launch_bounds__(256)
void dconv_drop_fallback(const float* __restrict__ x, const float* __restrict__ wgt,
                         const int* __restrict__ sidx, float* __restrict__ out)
{
  __shared__ int            s_idx[NT_ * KK_];
  __shared__ unsigned short s_b[2][32 * 72];
  const int t = threadIdx.x, lane = t & 31, wv = t >> 5;
  const int b = blockIdx.y, pix0 = blockIdx.x * NT_;
  for (int i = t; i < NT_ * KK_; i += 256) s_idx[i] = sidx[pix0 * KK_ + i] << 2;
  const char* xb = (const char*)(x + (size_t)b * CIN_ * HW_);
  const int m = wv & 3, npair = wv >> 2, koff = (lane < 16) ? 0 : 8;
  const float* wrow = wgt + (size_t)(m * 16 + (lane & 15)) * KTOT_;
  const int gkz = t >> 3, gpx = (t & 7) * 8;
  v8f a0v = {}, a1v = {};
  __syncthreads();
  for (int kc = 0, it = 0; kc < KTOT_; kc += 32, ++it) {
    unsigned short* sb = s_b[it & 1];
    {
      const int K = kc + gkz, c = K / KK_, kk = K - c * KK_;
      const char* xr = xb + (size_t)c * (HW_ * 4);
      float v[8];
#pragma unroll
      for (int i = 0; i < 8; ++i) v[i] = *(const float*)(xr + s_idx[(gpx + i) * KK_ + kk]);
      union { __bf16 h[8]; uint4 q; } P;
#pragma unroll
      for (int i = 0; i < 8; ++i) P.h[i] = (__bf16)v[i];
      *reinterpret_cast<uint4*>(&sb[gkz * 72 + gpx]) = P.q;
    }
    __syncthreads();
    union { v16bf v; uint4 q[2]; } A;
    {
      const float4 w0 = *(const float4*)(wrow + kc + koff);
      const float4 w1 = *(const float4*)(wrow + kc + koff + 4);
      const float4 w2 = *(const float4*)(wrow + kc + koff + 16);
      const float4 w3 = *(const float4*)(wrow + kc + koff + 20);
      union { __bf16 h[16]; uint4 q[2]; } U;
      U.h[0]=(__bf16)w0.x; U.h[1]=(__bf16)w0.y; U.h[2]=(__bf16)w0.z; U.h[3]=(__bf16)w0.w;
      U.h[4]=(__bf16)w1.x; U.h[5]=(__bf16)w1.y; U.h[6]=(__bf16)w1.z; U.h[7]=(__bf16)w1.w;
      U.h[8]=(__bf16)w2.x; U.h[9]=(__bf16)w2.y; U.h[10]=(__bf16)w2.z; U.h[11]=(__bf16)w2.w;
      U.h[12]=(__bf16)w3.x; U.h[13]=(__bf16)w3.y; U.h[14]=(__bf16)w3.z; U.h[15]=(__bf16)w3.w;
      A.q[0] = U.q[0]; A.q[1] = U.q[1];
    }
    union { v16bf v; uint4 q[2]; unsigned short u[16]; } B0, B1;
    const unsigned short* brow = &sb[npair * 32 + (lane & 15)];
#pragma unroll
    for (int s = 0; s < 16; ++s) {
      const int k = (s < 8) ? koff + s : 8 + koff + s;
      B0.u[s] = brow[k * 72]; B1.u[s] = brow[k * 72 + 16];
    }
    a0v = __builtin_amdgcn_wmma_f32_16x16x32_bf16(false, A.v, false, B0.v, (short)0, a0v, false, false);
    a1v = __builtin_amdgcn_wmma_f32_16x16x32_bf16(false, A.v, false, B1.v, (short)0, a1v, false, false);
  }
  const int ncol = lane & 15, moff = (lane < 16) ? 0 : 8;
  float* op = out + ((size_t)b * COUT_ + m * 16 + moff) * HW_ + pix0 + npair * 32 + ncol;
#pragma unroll
  for (int j = 0; j < 8; ++j) {
    op[(size_t)j * HW_]      = a0v[j];
    op[(size_t)j * HW_ + 16] = a1v[j];
  }
}

extern "C" void kernel_launch(void* const* d_in, const int* in_sizes, int n_in,
                              void* d_out, int out_size, void* d_ws, size_t ws_size,
                              hipStream_t stream) {
  const float* x   = (const float*)d_in[0];   // (32, 64, 56, 56) f32
  const float* w   = (const float*)d_in[1];   // (64, 64, 3, 3)   f32
  const int*   si  = (const int*)d_in[2];     // (56*56, 9)       i32
  float*       out = (float*)d_out;           // (32, 64, 56, 56) f32

  dim3 grid(HW_ / NT_, B_);                   // 49 pixel-tiles x 32 batches

  if (ws_size >= WPK_BYTES + XT_BYTES) {
    uint4*          wpk = (uint4*)d_ws;
    unsigned short* xt  = (unsigned short*)((char*)d_ws + WPK_BYTES);
    prepack_w_bfrag<<<(NCH_ * 4 * 32 + 255) / 256, 256, 0, stream>>>(w, wpk);
    transpose_x_bf16<<<grid, 256, 0, stream>>>(x, xt);
    dconv_drop_wmma_t<<<grid, 128, 0, stream>>>(xt, wpk, si, out);
  } else {
    dconv_drop_fallback<<<grid, 256, 0, stream>>>(x, w, si, out);
  }
}
